// PermutationPlackettLayer_24807731102181
// MI455X (gfx1250) — matrix operand order repaired and checked by reading, not compiled, not run on hardware
//
#include <hip/hip_runtime.h>
#include <cstdint>
#include <cstddef>

// ---------------------------------------------------------------------------
// PermutationPlackettLayer for MI455X (gfx1250, wave32, WMMA)
// B=32, N=1024, D=1024, H=512, action_refinement=1
// ---------------------------------------------------------------------------

typedef __attribute__((ext_vector_type(16))) _Float16 v16h;
typedef __attribute__((ext_vector_type(8)))  _Float16 v8h;
typedef __attribute__((ext_vector_type(8)))  float    v8f;
typedef int v4i_ __attribute__((vector_size(16)));

#define B_BATCH 32
#define SEQ_N   1024
#define DMODEL  1024
#define HID     512
#define M_TOTAL (B_BATCH * SEQ_N)   // 32768 rows

#define TEMP_C       1.0f
#define EULER_GAMMA  0.5772156649015329f
#define BIGC         1.0e9f

// LDS row strides (in halfs): 40 halfs = 80 bytes -> 16B-aligned rows,
// bank pattern 20*l mod 64 is distinct for 16 lanes (conflict-free b128 loads).
#define LDA 40
#define LDB 40

// workspace layout (bytes)
#define WS_W1T   0                                   // f16 [HID][DMODEL] = 1 MB
#define WS_S     (WS_W1T  + (size_t)HID * DMODEL * 2)
#define WS_EXPA  (WS_S    + (size_t)M_TOTAL * 4)
#define WS_APROD (WS_EXPA + (size_t)M_TOTAL * 4)
#define WS_BMAT  (WS_APROD+ (size_t)M_TOTAL * 4)
#define WS_PERMU (WS_BMAT + (size_t)M_TOTAL * 4)

// CDNA5 async global->LDS copy path (ASYNCcnt-tracked), guarded so the file
// still compiles on toolchains without the builtins (and on the host pass).
#if __has_builtin(__builtin_amdgcn_global_load_async_to_lds_b128) && \
    __has_builtin(__builtin_amdgcn_s_wait_asynccnt)
#define HAS_ASYNC_LDS 1
typedef __attribute__((address_space(1))) v4i_* gptr_v4i;
typedef __attribute__((address_space(3))) v4i_* lptr_v4i;
#else
#define HAS_ASYNC_LDS 0
#endif

// ---------------------------------------------------------------------------
// Kernel 0: W1 [D][H] f32 -> W1T [H][D] f16 via LDS tile transpose
// (coalesced both directions; result stays resident in 192MB L2, 1 MB)
// ---------------------------------------------------------------------------
__global__ __launch_bounds__(256) void cvt_w1t_kernel(
    const float* __restrict__ W1, _Float16* __restrict__ W1T)
{
    __shared__ _Float16 t[32][33];
    const int tid = threadIdx.x;
    const int bd  = blockIdx.x * 32;   // D tile base
    const int bh  = blockIdx.y * 32;   // H tile base
    #pragma unroll
    for (int i = 0; i < 4; ++i) {
        int e = tid + i * 256;
        int r = e >> 5, c = e & 31;
        t[r][c] = (_Float16)W1[(size_t)(bd + r) * HID + bh + c];  // coalesced in c
    }
    __syncthreads();
    #pragma unroll
    for (int i = 0; i < 4; ++i) {
        int e = tid + i * 256;
        int r = e >> 5, c = e & 31;
        W1T[(size_t)(bh + r) * DMODEL + bd + c] = t[c][r];        // coalesced in c
    }
}

// ---------------------------------------------------------------------------
// Kernel 1: fused  acts = relu(Q@W1 + b1) @ W2 + b2  -> softplus/log/noise
// One block = 16 rows x all 512 hidden columns. 8 waves x 4 WMMA tiles.
// ---------------------------------------------------------------------------
__global__ __launch_bounds__(256) void fused_mlp_wmma_kernel(
    const float* __restrict__ Q, const _Float16* __restrict__ W1T,
    const float* __restrict__ b1, const float* __restrict__ W2,
    const float* __restrict__ b2, const unsigned char* __restrict__ mask,
    const float* __restrict__ unoise, const float* __restrict__ gnoise,
    float* __restrict__ masked_out,   // d_out region [32,1024]
    float* __restrict__ s_ws,         // exp(masked+gumbel)*mask
    float* __restrict__ expa_ws,      // exp(masked)  (0 where masked)
    float* __restrict__ aprod_ws)     // acts * mask
{
    __shared__ __attribute__((aligned(16))) _Float16 As[16 * LDA];
    __shared__ __attribute__((aligned(16))) _Float16 Bs[HID * LDB];
    __shared__ float accLds[16];

    const int tid  = threadIdx.x;
    const int wave = tid >> 5;
    const int lane = tid & 31;
    const int m0   = blockIdx.x * 16;

    // A-tile loader mapping: each thread loads one float2 of Q
    const int am = tid >> 4;          // row 0..15
    const int ak = (tid & 15) * 2;    // k 0..30 even

    // fragment lane mapping (per CDNA5 ISA layouts)
    const int arow = lane & 15;
    const int akb  = (lane < 16) ? 0 : 16;   // A: K base 0 / 16 (16 contiguous K)
    const int bkb  = (lane < 16) ? 0 : 16;   // B: K base 0 / 16 (16 contiguous K)
    const int nc   = lane & 15;              // B column within tile

    v8f c[4];
    #pragma unroll
    for (int t = 0; t < 4; ++t) c[t] = (v8f){0.f,0.f,0.f,0.f,0.f,0.f,0.f,0.f};

    for (int kk = 0; kk < DMODEL; kk += 32) {
        __syncthreads();
        // ---- load A tile 16x32 f32 -> f16 (coalesced float2 per thread)
        const float2 q2 = *(const float2*)(Q + (size_t)(m0 + am) * DMODEL + kk + ak);
        As[am * LDA + ak]     = (_Float16)q2.x;
        As[am * LDA + ak + 1] = (_Float16)q2.y;
        if (kk + 32 < DMODEL)
            __builtin_prefetch(Q + (size_t)(m0 + am) * DMODEL + kk + 32 + ak, 0, 0);

        // ---- B tile: W1T[n][kk..kk+31] -> Bs[n*LDB + k], 512x32 halfs, b128
#if HAS_ASYNC_LDS
        #pragma unroll
        for (int i = 0; i < 8; ++i) {
            int e8 = tid + i * 256;                  // group of 8 halfs (16 B)
            int n = e8 >> 2, k8 = (e8 & 3) * 8;
            __builtin_amdgcn_global_load_async_to_lds_b128(
                (gptr_v4i)(W1T + (size_t)n * DMODEL + kk + k8),
                (lptr_v4i)(Bs + n * LDB + k8),
                0, 0);
        }
        __builtin_amdgcn_s_wait_asynccnt(0);
#else
        #pragma unroll
        for (int i = 0; i < 8; ++i) {
            int e8 = tid + i * 256;
            int n = e8 >> 2, k8 = (e8 & 3) * 8;
            *(v8h*)(Bs + n * LDB + k8) =
                *(const v8h*)(W1T + (size_t)n * DMODEL + kk + k8);
        }
#endif
        __syncthreads();

        // ---- A fragment: lanes 0-15: K0..15 ; lanes 16-31: K16..31
        v8h alo = *(const v8h*)(As + arow * LDA + akb);
        v8h ahi = *(const v8h*)(As + arow * LDA + akb + 8);
        v16h a  = __builtin_shufflevector(alo, ahi,
                    0,1,2,3,4,5,6,7,8,9,10,11,12,13,14,15);

        // ---- load all 4 B fragments first, then issue the 4 WMMAs,
        //      so ds loads overlap matrix issue (partial dscnt waits)
        v16h bf[4];
        #pragma unroll
        for (int t = 0; t < 4; ++t) {
            const _Float16* bp = Bs + (size_t)((wave * 4 + t) * 16 + nc) * LDB + bkb;
            v8h blo = *(const v8h*)bp;
            v8h bhi = *(const v8h*)(bp + 8);
            bf[t] = __builtin_shufflevector(blo, bhi,
                        0,1,2,3,4,5,6,7,8,9,10,11,12,13,14,15);
        }
        #pragma unroll
        for (int t = 0; t < 4; ++t) {
            c[t] = __builtin_amdgcn_wmma_f32_16x16x32_f16(
                       false, a, false, bf[t], (short)0, c[t], false, false);
        }
    }

    // ---- epilogue: relu(c + b1) * W2, reduce over all 512 columns per row
    if (tid < 16) accLds[tid] = 0.f;
    __syncthreads();

    #pragma unroll
    for (int t = 0; t < 4; ++t) {
        int n = (wave * 4 + t) * 16 + nc;
        float b1v = b1[n];
        float w2v = W2[n];
        #pragma unroll
        for (int r = 0; r < 8; ++r) {
            float v = fmaxf(c[t][r] + b1v, 0.f) * w2v;
            // sum over the 16 lanes of this half-wave (one C-tile row)
            v += __shfl_xor(v, 1);
            v += __shfl_xor(v, 2);
            v += __shfl_xor(v, 4);
            v += __shfl_xor(v, 8);
            if ((lane & 15) == 0) {
                int rowl = r + ((lane >> 4) << 3);   // lanes16-31 -> rows 8..15
                atomicAdd(&accLds[rowl], v);
            }
        }
    }
    __syncthreads();

    // ---- per-row scalar tail (softplus clamp + log + noise, masks, exps)
    if (tid < 16) {
        int row = m0 + tid;
        float araw = accLds[tid] + b2[0];
        float sp   = (araw > 0.f) ? (araw + log1pf(expf(-araw)))
                                  : log1pf(expf(araw));
        float acts = logf(fmaxf(sp, 1e-5f)) + unoise[row];
        float mk   = mask[row] ? 1.f : 0.f;
        float masked = (mk > 0.f) ? acts : -1.0e6f;
        masked_out[row] = masked;
        expa_ws[row]    = (mk > 0.f) ? expf(acts) : 0.f;   // exp(-1e6) == 0
        s_ws[row]       = mk * expf(masked + gnoise[row]); // NeuralSort scores
        aprod_ws[row]   = acts * mk;                       // for KL terms
    }
}

// ---------------------------------------------------------------------------
// Kernel 2: Bmat[b][j] = sum_k |s_j - s_k|  (one block per batch)
// ---------------------------------------------------------------------------
__global__ __launch_bounds__(256) void bmat_kernel(
    const float* __restrict__ s_ws, float* __restrict__ Bmat)
{
    const int b = blockIdx.x, tid = threadIdx.x;
    __shared__ float sS[SEQ_N];
    for (int t = 0; t < 4; ++t) {
        int j = tid + t * 256;
        sS[j] = s_ws[b * SEQ_N + j];
    }
    __syncthreads();
    for (int t = 0; t < 4; ++t) {
        int j = tid + t * 256;
        float sj = sS[j];
        float acc = 0.f;
        for (int k = 0; k < SEQ_N; ++k) acc += fabsf(sj - sS[k]);
        Bmat[b * SEQ_N + j] = acc;
    }
}

// ---------------------------------------------------------------------------
// Kernel 3: greedy argmax permutation + Plackett-Luce log-lik + KL
// (one block per batch; sequential over N rows, parallel argmax per row)
// ---------------------------------------------------------------------------
__global__ __launch_bounds__(256) void permute_pl_kl_kernel(
    const float* __restrict__ s_ws, const float* __restrict__ Bmat,
    const float* __restrict__ expa, const float* __restrict__ aprod,
    const unsigned char* __restrict__ mask,
    int* __restrict__ permu_out,
    float* __restrict__ kl_out, float* __restrict__ lognom_out,
    float* __restrict__ lognorm_out)
{
    const int b = blockIdx.x, tid = threadIdx.x;
    __shared__ float sS[SEQ_N], sB[SEQ_N], sFlag[SEQ_N], sM[SEQ_N];
    __shared__ float sEx[SEQ_N], sSuf[SEQ_N];
    __shared__ int   sChosen[SEQ_N];
    __shared__ float redV[256];
    __shared__ int   redI[256];

    for (int t = 0; t < 4; ++t) {
        int i = tid + t * 256;
        sS[i]    = s_ws[b * SEQ_N + i];
        sB[i]    = Bmat[b * SEQ_N + i];
        sM[i]    = mask[b * SEQ_N + i] ? 1.f : 0.f;
        sFlag[i] = 0.f;
    }
    __syncthreads();

    // greedy row-by-row argmax of P_hat[b,i,j] = s_j*scale_i - B_j - penalties
    for (int i = 0; i < SEQ_N; ++i) {
        float scale = (float)(SEQ_N - 1 - 2 * i);
        float bv = -3.4e38f; int bi = 0;
        #pragma unroll
        for (int t = 0; t < 4; ++t) {
            int j = tid + t * 256;
            float v = sS[j] * scale - sB[j]
                      - (sFlag[j] + (1.f - sM[j])) * BIGC;
            if (v > bv) { bv = v; bi = j; }
        }
        redV[tid] = bv; redI[tid] = bi;
        __syncthreads();
        for (int off = 128; off > 0; off >>= 1) {
            if (tid < off) {
                float v2 = redV[tid + off]; int i2 = redI[tid + off];
                if (v2 > redV[tid] || (v2 == redV[tid] && i2 < redI[tid])) {
                    redV[tid] = v2; redI[tid] = i2;
                }
            }
            __syncthreads();
        }
        if (tid == 0) { int idx = redI[0]; sChosen[i] = idx; sFlag[idx] = 1.f; }
        __syncthreads();
    }

    // shift (prepend 0, drop last), mask -> identity; gather exp(masked_acts)
    for (int t = 0; t < 4; ++t) {
        int i = tid + t * 256;
        int pm = (i == 0) ? 0 : sChosen[i - 1];
        if (!(sM[i] > 0.f)) pm = i;
        permu_out[b * SEQ_N + i] = pm;
        float e = expa[b * SEQ_N + pm];
        sEx[i] = e; sSuf[i] = e;
    }
    __syncthreads();

    // inclusive suffix cumsum (Hillis-Steele), 10 doubling steps
    for (int d = 1; d < SEQ_N; d <<= 1) {
        float add[4];
        #pragma unroll
        for (int t = 0; t < 4; ++t) {
            int i = tid + t * 256;
            add[t] = (i + d < SEQ_N) ? sSuf[i + d] : 0.f;
        }
        __syncthreads();
        #pragma unroll
        for (int t = 0; t < 4; ++t) sSuf[tid + t * 256] += add[t];
        __syncthreads();
    }

    // per-thread partial sums: PL term, KL s1/s2, n_valid
    float tsum = 0.f, s1 = 0.f, s2 = 0.f, nv = 0.f;
    for (int t = 0; t < 4; ++t) {
        int i = tid + t * 256;
        float mf = sM[i];
        tsum += (logf(sEx[i] + 1e-20f) - logf(sSuf[i] + 1e-20f)) * mf;
        float ap = aprod[b * SEQ_N + i];
        s1 += (1.0f / TEMP_C) * ap;
        s2 += expf(-(1.0f / TEMP_C) * fmaxf(ap, -20.0f * TEMP_C));
        nv += mf;
    }
    // four block reductions
    float sums[4] = {tsum, s1, s2, nv};
    float res[4];
    for (int q = 0; q < 4; ++q) {
        redV[tid] = sums[q];
        __syncthreads();
        for (int off = 128; off > 0; off >>= 1) {
            if (tid < off) redV[tid] += redV[tid + off];
            __syncthreads();
        }
        res[q] = redV[0];
        __syncthreads();
    }
    if (tid == 0) {
        float n_valid = res[3];
        float kl1 = n_valid * (logf(TEMP_C) - 1.0f
                    + EULER_GAMMA * (1.0f / TEMP_C - 1.0f));
        float s2f = res[2] - ((float)SEQ_N - n_valid);
        float kl  = kl1 + res[1] + s2f * expf(lgammaf(1.0f + 1.0f / TEMP_C));
        kl_out[b]      = kl;
        lognom_out[b]  = res[0];
        lognorm_out[b] = 0.f;
    }
}

// ---------------------------------------------------------------------------
// Kernel 4: one-hot permutation matrix, pure 134 MB streaming b128 store
// ---------------------------------------------------------------------------
__global__ __launch_bounds__(256) void onehot_kernel(
    const int* __restrict__ permu, float* __restrict__ out)
{
    int row = blockIdx.x;                 // 0 .. M_TOTAL-1  == (b, i)
    int p   = permu[row];
    float4* o = (float4*)(out + (size_t)row * SEQ_N);
    int j0 = threadIdx.x * 4;
    float4 v;
    v.x = (j0     == p) ? 1.f : 0.f;
    v.y = (j0 + 1 == p) ? 1.f : 0.f;
    v.z = (j0 + 2 == p) ? 1.f : 0.f;
    v.w = (j0 + 3 == p) ? 1.f : 0.f;
    o[threadIdx.x] = v;
}

// ---------------------------------------------------------------------------
extern "C" void kernel_launch(void* const* d_in, const int* in_sizes, int n_in,
                              void* d_out, int out_size, void* d_ws, size_t ws_size,
                              hipStream_t stream) {
    (void)in_sizes; (void)n_in; (void)out_size; (void)ws_size;

    const float*         Q    = (const float*)d_in[0];
    const unsigned char* mask = (const unsigned char*)d_in[1];
    const float*         W1   = (const float*)d_in[2];
    const float*         b1   = (const float*)d_in[3];
    const float*         W2   = (const float*)d_in[4];
    const float*         b2   = (const float*)d_in[5];
    const float*         un   = (const float*)d_in[6];
    const float*         gn   = (const float*)d_in[7];
    // d_in[8] = action_refinement (== 1, ignored)

    float* out = (float*)d_out;
    float* out_permu_mat = out;                                   // [32,1024,1024]
    float* out_masked    = out + (size_t)M_TOTAL * SEQ_N;         // [32,1024]
    float* out_kl        = out_masked + M_TOTAL;                  // [32,1]
    float* out_lognom    = out_kl + B_BATCH;                      // [32,1]
    float* out_lognorm   = out_lognom + B_BATCH;                  // [32,1]

    char* ws = (char*)d_ws;
    _Float16* w1t   = (_Float16*)(ws + WS_W1T);
    float*    s_ws  = (float*)(ws + WS_S);
    float*    expa  = (float*)(ws + WS_EXPA);
    float*    aprod = (float*)(ws + WS_APROD);
    float*    bmat  = (float*)(ws + WS_BMAT);
    int*      permu = (int*)(ws + WS_PERMU);

    // 0) W1 -> W1T f16 (1 MB, L2-resident for all GEMM blocks)
    cvt_w1t_kernel<<<dim3(DMODEL / 32, HID / 32), 256, 0, stream>>>(W1, w1t);

    // 1) fused WMMA MLP: 2048 blocks of 16 rows, full H per block
    fused_mlp_wmma_kernel<<<M_TOTAL / 16, 256, 0, stream>>>(
        Q, w1t, b1, W2, b2, mask, un, gn, out_masked, s_ws, expa, aprod);

    // 2) NeuralSort row sums
    bmat_kernel<<<B_BATCH, 256, 0, stream>>>(s_ws, bmat);

    // 3) greedy permutation + PL log-likelihood + KL
    permute_pl_kl_kernel<<<B_BATCH, 256, 0, stream>>>(
        s_ws, bmat, expa, aprod, mask, permu, out_kl, out_lognom, out_lognorm);

    // 4) one-hot output (134 MB stream)
    onehot_kernel<<<M_TOTAL, 256, 0, stream>>>(permu, out_permu_mat);
}
